// SDENet_39470749450705
// MI455X (gfx1250) — matrix operand order repaired and checked by reading, not compiled, not run on hardware
//
#include <hip/hip_runtime.h>
#include <math.h>

typedef __attribute__((ext_vector_type(16))) _Float16 v16h;
typedef __attribute__((ext_vector_type(8)))  _Float16 v8h;
typedef __attribute__((ext_vector_type(8)))  float    v8f;

#define B_TOT 131072
#define IN_D  90
#define H_D   50
#define DH_D  100
#define DEPTH 10
#define INP   96    // 90 padded to 3 k-tiles of 32
#define HP    64    // 50 padded to 2 k-tiles / 4 n-tiles
#define DHP   112   // 100 padded to 7 n-tiles
#define NW    4     // waves per block
#define TPB   (NW * 32)

// ---- WMMA fragment loaders (f16 16x16x32 layouts, wave32) ----
// A (16x32, M x K): lane l holds row m=l&15; k = 8*hi + (j<8 ? j : j+8)
// -> two contiguous 16B LDS reads per k-tile.
__device__ __forceinline__ v16h load_a_frag(const _Float16* panel, int stride,
                                            int k0, int m, int hi) {
  const v8h* p = (const v8h*)(panel + m * stride + k0 + hi * 8);
  v8h lo = p[0];
  v8h hh = p[2];  // +16 halfs
  v16h a;
#pragma unroll
  for (int j = 0; j < 8; ++j) { a[j] = lo[j]; a[j + 8] = hh[j]; }
  return a;
}

// B (32x16, K x N): lane l holds col n=l&15; k = 8*hi + (j<8 ? j : j+8).
// Weights stored transposed (n-major) in LDS -> two contiguous 8-half (16B) reads.
__device__ __forceinline__ v16h load_b_frag(const _Float16* wT, int strideK,
                                            int k0, int n, int hi) {
  const v8h* p = (const v8h*)(wT + n * strideK + k0 + hi * 8);
  v8h b0 = p[0];
  v8h b1 = p[2];
  v16h b;
#pragma unroll
  for (int j = 0; j < 8; ++j) { b[j] = b0[j]; b[j + 8] = b1[j]; }
  return b;
}

__global__ __launch_bounds__(TPB) void sdenet_fused_kernel(
    const float* __restrict__ x, const float* __restrict__ noise,
    const float* __restrict__ W_down, const float* __restrict__ b_down,
    const float* __restrict__ W_drift, const float* __restrict__ b_drift,
    const float* __restrict__ W_d1, const float* __restrict__ b_d1,
    const float* __restrict__ W_d2, const float* __restrict__ b_d2,
    const float* __restrict__ W_fc, const float* __restrict__ b_fc,
    float* __restrict__ out_mean, float* __restrict__ out_sigma) {
  // Weights, transposed + zero-padded, as f16
  __shared__ __align__(16) _Float16 sWdownT[HP * INP];   // [n][k]
  __shared__ __align__(16) _Float16 sWdriftT[HP * HP];   // [n][k]
  __shared__ __align__(16) _Float16 sWd1T[DHP * HP];     // [n][k]
  __shared__ float sWd2[DHP];
  __shared__ float sBdown[HP], sBdrift[HP], sBd1[DHP];
  __shared__ float sWfc0[HP], sWfc1[HP];
  __shared__ float sMisc[4];  // [0]=b_fc0 [1]=b_fc1 [2]=b_d2
  // Per-wave activation panels (wave-private; no cross-wave traffic)
  __shared__ __align__(16) _Float16 sX[NW][16 * INP];
  __shared__ __align__(16) _Float16 sOut[NW][16 * HP];

  const int tid  = threadIdx.x;
  const int wave = tid >> 5;
  const int lane = tid & 31;
  const int m16  = lane & 15;   // A-row / B-col / C-col index
  const int hi   = lane >> 4;   // lane-half

  // ---- Cooperative weight load (once) ----
  for (int i = tid; i < HP * INP; i += TPB) {
    int n = i / INP, k = i % INP;
    sWdownT[i] = (_Float16)((k < IN_D && n < H_D) ? W_down[k * H_D + n] : 0.f);
  }
  for (int i = tid; i < HP * HP; i += TPB) {
    int n = i / HP, k = i % HP;
    sWdriftT[i] = (_Float16)((k < H_D && n < H_D) ? W_drift[k * H_D + n] : 0.f);
  }
  for (int i = tid; i < DHP * HP; i += TPB) {
    int n = i / HP, k = i % HP;
    sWd1T[i] = (_Float16)((k < H_D && n < DH_D) ? W_d1[k * DH_D + n] : 0.f);
  }
  for (int i = tid; i < DHP; i += TPB) {
    sWd2[i] = (i < DH_D) ? W_d2[i] : 0.f;
    sBd1[i] = (i < DH_D) ? b_d1[i] : 0.f;
  }
  for (int i = tid; i < HP; i += TPB) {
    sBdown[i]  = (i < H_D) ? b_down[i] : 0.f;
    sBdrift[i] = (i < H_D) ? b_drift[i] : 0.f;
    sWfc0[i]   = (i < H_D) ? W_fc[i * 2 + 0] : 0.f;
    sWfc1[i]   = (i < H_D) ? W_fc[i * 2 + 1] : 0.f;
  }
  if (tid == 0) { sMisc[0] = b_fc[0]; sMisc[1] = b_fc[1]; sMisc[2] = b_d2[0]; }

  // ---- Stage this wave's 16-row x tile (coalesced, f32 -> f16) ----
  const int rowBase = blockIdx.x * (NW * 16) + wave * 16;
  _Float16* xp = &sX[wave][0];
  for (int i = lane; i < 16 * INP; i += 32) {
    int r = i / INP, c = i % INP;
    xp[i] = (_Float16)((c < IN_D) ? x[(size_t)(rowBase + r) * IN_D + c] : 0.f);
  }
  // warm the cache for the first SDE step's noise tile (contiguous 3.2KB)
  {
    const char* np = (const char*)(noise + (size_t)rowBase * H_D);
    __builtin_prefetch(np + lane * 128, 0, 1);
  }
  __syncthreads();

  _Float16* op = &sOut[wave][0];

  // ---- out = x @ W_down + b_down  (16x96 * 96x64, 12 WMMAs) ----
  v8f co[4];
#pragma unroll
  for (int nt = 0; nt < 4; ++nt) {
    v8f c = {};
#pragma unroll
    for (int kt = 0; kt < 3; ++kt) {
      v16h a = load_a_frag(xp, INP, kt * 32, m16, hi);
      v16h b = load_b_frag(sWdownT, INP, kt * 32, nt * 16 + m16, hi);
      c = __builtin_amdgcn_wmma_f32_16x16x32_f16(false, a, false, b,
                                                 (short)0, c, false, false);
    }
    float bb = sBdown[nt * 16 + m16];
#pragma unroll
    for (int v = 0; v < 8; ++v) c[v] += bb;
    co[nt] = c;
  }

  // spill out-state to the wave panel (A-source for d1 GEMM)
#pragma unroll
  for (int nt = 0; nt < 4; ++nt)
#pragma unroll
    for (int v = 0; v < 8; ++v)
      op[(v + 8 * hi) * HP + nt * 16 + m16] = (_Float16)co[nt][v];

  // ---- diffusion = 0.5*sigmoid(relu(out@W_d1+b)@W_d2 + b_d2) (14 WMMAs) ----
  float part[8];
#pragma unroll
  for (int v = 0; v < 8; ++v) part[v] = 0.f;
#pragma unroll
  for (int nt = 0; nt < 7; ++nt) {
    v8f c = {};
#pragma unroll
    for (int kt = 0; kt < 2; ++kt) {
      v16h a = load_a_frag(op, HP, kt * 32, m16, hi);
      v16h b = load_b_frag(sWd1T, HP, kt * 32, nt * 16 + m16, hi);
      c = __builtin_amdgcn_wmma_f32_16x16x32_f16(false, a, false, b,
                                                 (short)0, c, false, false);
    }
    int n = nt * 16 + m16;
    float bb = sBd1[n], w2 = sWd2[n];
#pragma unroll
    for (int v = 0; v < 8; ++v) part[v] += fmaxf(c[v] + bb, 0.f) * w2;
  }
  // reduce across the 16 lanes of each half (columns of h)
#pragma unroll
  for (int off = 1; off < 16; off <<= 1)
#pragma unroll
    for (int v = 0; v < 8; ++v) part[v] += __shfl_xor(part[v], off, 32);
  float diffv[8];
  {
    float bd2v = sMisc[2];
#pragma unroll
    for (int v = 0; v < 8; ++v)
      diffv[v] = 0.5f / (1.f + expf(-(part[v] + bd2v)));  // row m = v + 8*hi
  }

  // ---- Hoist loop-invariant W_drift B-fragments into registers (64 VGPRs) ----
  v16h bdrift_frag[4][2];
#pragma unroll
  for (int nt = 0; nt < 4; ++nt)
#pragma unroll
    for (int kt = 0; kt < 2; ++kt)
      bdrift_frag[nt][kt] = load_b_frag(sWdriftT, HP, kt * 32, nt * 16 + m16, hi);

  // ---- SDE scan: 10 steps of out += relu(out@W_drift+b)*dt + diff*sdt*noise ----
  const float dt = 0.4f, sdt = 0.63245553203f;
  for (int step = 0; step < DEPTH; ++step) {
    // prefetch next step's contiguous 16x50 f32 noise tile across the serial dep
    if (step + 1 < DEPTH) {
      const char* np =
          (const char*)(noise + ((size_t)(step + 1) * B_TOT + rowBase) * H_D);
      __builtin_prefetch(np + lane * 128, 0, 1);
    }

    // refresh wave-private A panel from fp32 fragments
#pragma unroll
    for (int nt = 0; nt < 4; ++nt)
#pragma unroll
      for (int v = 0; v < 8; ++v)
        op[(v + 8 * hi) * HP + nt * 16 + m16] = (_Float16)co[nt][v];

#pragma unroll
    for (int nt = 0; nt < 4; ++nt) {
      v8f c = {};
#pragma unroll
      for (int kt = 0; kt < 2; ++kt) {
        v16h a = load_a_frag(op, HP, kt * 32, m16, hi);
        c = __builtin_amdgcn_wmma_f32_16x16x32_f16(false, a, false,
                                                   bdrift_frag[nt][kt],
                                                   (short)0, c, false, false);
      }
      int n = nt * 16 + m16;
      float bb = sBdrift[n];
      bool valid = (n < H_D);
#pragma unroll
      for (int v = 0; v < 8; ++v) {
        float drift = fmaxf(c[v] + bb, 0.f);
        int row = rowBase + v + 8 * hi;
        float nz = valid
            ? noise[((size_t)step * B_TOT + row) * H_D + n]
            : 0.f;
        co[nt][v] += drift * dt + diffv[v] * sdt * nz;
      }
    }
  }

  // ---- final = relu(out) @ W_fc + b_fc ; mean / softplus sigma ----
  float p0[8], p1[8];
#pragma unroll
  for (int v = 0; v < 8; ++v) { p0[v] = 0.f; p1[v] = 0.f; }
#pragma unroll
  for (int nt = 0; nt < 4; ++nt) {
    int n = nt * 16 + m16;
    float w0 = sWfc0[n], w1 = sWfc1[n];
#pragma unroll
    for (int v = 0; v < 8; ++v) {
      float r = fmaxf(co[nt][v], 0.f);
      p0[v] += r * w0;
      p1[v] += r * w1;
    }
  }
#pragma unroll
  for (int off = 1; off < 16; off <<= 1)
#pragma unroll
    for (int v = 0; v < 8; ++v) {
      p0[v] += __shfl_xor(p0[v], off, 32);
      p1[v] += __shfl_xor(p1[v], off, 32);
    }
  if (m16 == 0) {
    float bfc0 = sMisc[0], bfc1 = sMisc[1];
#pragma unroll
    for (int v = 0; v < 8; ++v) {
      int row = rowBase + v + 8 * hi;
      float mean = p0[v] + bfc0;
      float s = p1[v] + bfc1;
      float sp = fmaxf(s, 0.f) + log1pf(expf(-fabsf(s)));  // stable softplus
      out_mean[row]  = mean;
      out_sigma[row] = sp + 0.001f;
    }
  }
}

extern "C" void kernel_launch(void* const* d_in, const int* in_sizes, int n_in,
                              void* d_out, int out_size, void* d_ws, size_t ws_size,
                              hipStream_t stream) {
  const float* x       = (const float*)d_in[0];
  const float* noise   = (const float*)d_in[1];
  const float* W_down  = (const float*)d_in[2];
  const float* b_down  = (const float*)d_in[3];
  const float* W_drift = (const float*)d_in[4];
  const float* b_drift = (const float*)d_in[5];
  const float* W_d1    = (const float*)d_in[6];
  const float* b_d1    = (const float*)d_in[7];
  const float* W_d2    = (const float*)d_in[8];
  const float* b_d2    = (const float*)d_in[9];
  const float* W_fc    = (const float*)d_in[10];
  const float* b_fc    = (const float*)d_in[11];
  float* out = (float*)d_out;

  dim3 grid(B_TOT / (NW * 16));  // 2048 blocks x 128 threads (4 wave32)
  sdenet_fused_kernel<<<grid, TPB, 0, stream>>>(
      x, noise, W_down, b_down, W_drift, b_drift, W_d1, b_d1, W_d2, b_d2,
      W_fc, b_fc, out /*mean*/, out + B_TOT /*sigma*/);
}
